// GPTClosedLoop_8057358647376
// MI455X (gfx1250) — matrix operand order repaired and checked by reading, not compiled, not run on hardware
//
#include <hip/hip_runtime.h>
#include <hip/hip_bf16.h>
#include <math.h>

// Problem constants (from reference)
#define BATCH   64
#define TSTEPS  128
#define NREF    2
#define LAYERS  8
#define HEADS   8
#define DMODEL  512
#define HDIM    64
#define TMAXC   129

typedef __attribute__((ext_vector_type(16))) __bf16 v16bf;
typedef __attribute__((ext_vector_type(8)))  float  v8f;
typedef unsigned int u32x4 __attribute__((ext_vector_type(4)));
typedef int          i32x4 __attribute__((ext_vector_type(4)));
typedef int          i32x8 __attribute__((ext_vector_type(8)));

#if defined(__HIP_DEVICE_COMPILE__) && __has_builtin(__builtin_amdgcn_tensor_load_to_lds)
#define USE_TDM 1
#else
#define USE_TDM 0
#endif

struct U32x8 { uint4 lo, hi; };

__device__ __forceinline__ unsigned short f2bf(float f) {
    unsigned int u = __builtin_bit_cast(unsigned int, f);
    u += 0x7FFFu + ((u >> 16) & 1u);           // round-to-nearest-even
    return (unsigned short)(u >> 16);
}
__device__ __forceinline__ float bf2f(unsigned short h) {
    unsigned int u = ((unsigned int)h) << 16;
    return __builtin_bit_cast(float, u);
}
// 16 contiguous bf16 (32B) -> v16bf
__device__ __forceinline__ v16bf load_frag16(const unsigned short* p) {
    U32x8 r; r.lo = *(const uint4*)p; r.hi = *(const uint4*)(p + 8);
    return __builtin_bit_cast(v16bf, r);
}
// A fragment: 8 bf16 at p, 8 bf16 at p+16 (K and K+16 halves per ISA layout)
__device__ __forceinline__ v16bf load_fragA(const unsigned short* p) {
    U32x8 r; r.lo = *(const uint4*)p; r.hi = *(const uint4*)(p + 16);
    return __builtin_bit_cast(v16bf, r);
}

// ---------------------------------------------------------------------------
// Weight pack: fp32 [L][K][N] row-major -> bf16 WMMA-B layout
// P[l][ntile][kchunk][lane][i]; lane 0-15: N=ntile*16+lane, K=kc*32+i (i=0..15)
//                              lane16-31: same N, K=kc*32+16+i
// ---------------------------------------------------------------------------
__global__ void pack_weights(const float* __restrict__ W,
                             unsigned short* __restrict__ P, int K, int N) {
    size_t KN = (size_t)K * N;
    size_t total = (size_t)LAYERS * KN;
    int kchunks = K / 32;
    for (size_t idx = (size_t)blockIdx.x * blockDim.x + threadIdx.x; idx < total;
         idx += (size_t)gridDim.x * blockDim.x) {
        size_t l   = idx / KN;
        size_t rem = idx - l * KN;
        int i  = (int)(rem & 15);
        int ln = (int)((rem >> 4) & 31);
        int kc = (int)((rem >> 9) % kchunks);
        int nt = (int)(rem / (512ull * kchunks));
        int n = nt * 16 + (ln & 15);
        int k = kc * 32 + (ln >> 4) * 16 + i;
        P[idx] = f2bf(W[l * KN + (size_t)k * N + n]);
    }
}

__global__ void init_state(const float* __restrict__ y_d,
                           float* __restrict__ y, float* __restrict__ u) {
    int i = blockIdx.x * blockDim.x + threadIdx.x;
    if (i < BATCH * NREF) {
        int b = i >> 1, nr = i & 1;
        y[i] = y_d[((size_t)b * TSTEPS + 0) * NREF + nr];
        u[i] = (nr == 0) ? 191.713f : 215.888f;
    }
}

// ---------------------------------------------------------------------------
// Embedding: tok=[r_t - y, u] @ wte_w + wte_b + wpe[t];  also emit Y[:,t]=y
// ---------------------------------------------------------------------------
__global__ void embed_kernel(const float* __restrict__ r,
                             const float* __restrict__ wte_w,
                             const float* __restrict__ wte_b,
                             const float* __restrict__ wpe,
                             const float* __restrict__ y,
                             const float* __restrict__ u,
                             float* __restrict__ h,
                             float* __restrict__ Y, int t) {
    int b = blockIdx.x;
    float e0 = r[((size_t)b * TSTEPS + t) * NREF + 0] - y[b * NREF + 0];
    float e1 = r[((size_t)b * TSTEPS + t) * NREF + 1] - y[b * NREF + 1];
    float u0 = u[b * NREF + 0], u1 = u[b * NREF + 1];
    for (int d = threadIdx.x; d < DMODEL; d += blockDim.x) {
        float v = e0 * wte_w[0 * DMODEL + d] + e1 * wte_w[1 * DMODEL + d]
                + u0 * wte_w[2 * DMODEL + d] + u1 * wte_w[3 * DMODEL + d]
                + wte_b[d] + wpe[(size_t)t * DMODEL + d];
        h[b * DMODEL + d] = v;
    }
    if (threadIdx.x < NREF)
        Y[((size_t)b * TSTEPS + t) * NREF + threadIdx.x] = y[b * NREF + threadIdx.x];
}

// ---------------------------------------------------------------------------
// Fused GEMM: C(64xN) = A(64xK) @ Wpacked(KxN) [+bias] with epilogue
//   MODE 0: QKV  -> q to qbuf (f32), k/v scattered to bf16 KV cache at pos
//   MODE 1: ADD  -> h += C + bias (residual)
//   MODE 2: GELU -> mout = bf16(gelu(C + bias))
// DOLN: LayerNorm(h) fused into the LDS A-staging phase (vectorized).
// Non-LN A staging uses the Tensor Data Mover (2D tile 64 x 512 bf16).
// Block = 8 waves; each wave: 16 columns x all 64 rows (4 WMMA acc tiles).
// ---------------------------------------------------------------------------
template <int KTOT, int MODE, bool DOLN>
__global__ __launch_bounds__(256) void gemm64_wmma(
    const float* __restrict__ Afp,            // h (DOLN path)
    const unsigned short* __restrict__ Abf,   // bf16 activations (non-LN path)
    const unsigned short* __restrict__ Wp,    // packed bf16 weights
    const float* __restrict__ bias,
    const float* __restrict__ lnw, const float* __restrict__ lnb,
    float* __restrict__ hio,                  // MODE 1
    float* __restrict__ qbuf,                 // MODE 0
    unsigned short* __restrict__ Kc,          // MODE 0
    unsigned short* __restrict__ Vc,          // MODE 0
    unsigned short* __restrict__ mout,        // MODE 2
    int layer, int pos, int Ntot) {
    __shared__ __align__(16) unsigned short Atile[64 * 512];

    const int wave   = threadIdx.x >> 5;
    const int lane   = threadIdx.x & 31;
    const int n0     = blockIdx.x * 128 + wave * 16;
    const int hiHalf = lane >> 4;    // 0: K 0..15, 1: K 16..31
    const int nloc   = lane & 15;

    v8f acc[4];
#pragma unroll
    for (int mt = 0; mt < 4; ++mt)
#pragma unroll
        for (int j = 0; j < 8; ++j) acc[mt][j] = 0.0f;

    constexpr int KB = KTOT / 512;
    for (int kb = 0; kb < KB; ++kb) {
        if (DOLN) {
            // fused LayerNorm: lane owns 16 contiguous elems; b128 LDS stores
            for (int rr = 0; rr < 8; ++rr) {
                int row = wave * 8 + rr;
                const float4* src = (const float4*)&Afp[row * 512];
                float4 v[4];
                float s = 0.0f, q = 0.0f;
#pragma unroll
                for (int c = 0; c < 4; ++c) {
                    v[c] = src[lane * 4 + c];
                    s += v[c].x + v[c].y + v[c].z + v[c].w;
                    q += v[c].x * v[c].x + v[c].y * v[c].y
                       + v[c].z * v[c].z + v[c].w * v[c].w;
                }
#pragma unroll
                for (int off = 16; off; off >>= 1) {
                    s += __shfl_xor(s, off, 32);
                    q += __shfl_xor(q, off, 32);
                }
                float mu  = s * (1.0f / 512.0f);
                float var = q * (1.0f / 512.0f) - mu * mu;
                float rsv = rsqrtf(var + 1e-5f);
                const float4* wv = (const float4*)lnw;
                const float4* bv = (const float4*)lnb;
                unsigned out[8];
#pragma unroll
                for (int c = 0; c < 4; ++c) {
                    float4 w4 = wv[lane * 4 + c];
                    float4 b4 = bv[lane * 4 + c];
                    float x0 = (v[c].x - mu) * rsv * w4.x + b4.x;
                    float x1 = (v[c].y - mu) * rsv * w4.y + b4.y;
                    float x2 = (v[c].z - mu) * rsv * w4.z + b4.z;
                    float x3 = (v[c].w - mu) * rsv * w4.w + b4.w;
                    out[c * 2 + 0] = (unsigned)f2bf(x0) | ((unsigned)f2bf(x1) << 16);
                    out[c * 2 + 1] = (unsigned)f2bf(x2) | ((unsigned)f2bf(x3) << 16);
                }
                uint4* dst = (uint4*)&Atile[row * 512 + lane * 16];
                dst[0] = make_uint4(out[0], out[1], out[2], out[3]);
                dst[1] = make_uint4(out[4], out[5], out[6], out[7]);
            }
        } else {
#if USE_TDM
            // Tensor Data Mover: 2D tile (64 rows x 512 bf16), row stride KTOT
            if (threadIdx.x == 0) {
                unsigned long long ga =
                    (unsigned long long)(const void*)Abf + (unsigned long long)kb * 512ull * 2ull;
                unsigned ldsa = (unsigned)(unsigned long long)(void*)&Atile[0];
                u32x4 g0;
                g0[0] = 1u;                                   // count=1 (valid D#)
                g0[1] = ldsa;                                 // lds_addr
                g0[2] = (unsigned)ga;                         // global_addr[31:0]
                g0[3] = ((unsigned)(ga >> 32) & 0x01FFFFFFu)  // global_addr[56:32]
                      | 0x80000000u;                          // type=2 ("image")
                i32x8 g1;
                g1[0] = 1 << 16;                  // data_size=2B
                g1[1] = (KTOT & 0xFFFF) << 16;    // tensor_dim0[15:0]
                g1[2] = (KTOT >> 16) | (64 << 16);// tensor_dim0[31:16] | tensor_dim1[15:0]
                g1[3] = 512 << 16;                // tensor_dim1[31:16]=0 | tile_dim0=512
                g1[4] = 64;                       // tile_dim1=64, tile_dim2=0
                g1[5] = KTOT;                     // tensor_dim0_stride[31:0]
                g1[6] = 0;                        // stride0[47:32] | dim1_stride[15:0]
                g1[7] = 0;
                i32x4 gz; gz[0] = 0; gz[1] = 0; gz[2] = 0; gz[3] = 0;
                i32x8 gz8;
                gz8[0] = 0; gz8[1] = 0; gz8[2] = 0; gz8[3] = 0;
                gz8[4] = 0; gz8[5] = 0; gz8[6] = 0; gz8[7] = 0;
                __builtin_amdgcn_tensor_load_to_lds(g0, g1, gz, gz, gz8, 0);
                __builtin_amdgcn_s_wait_tensorcnt(0);
            }
#else
            for (int idx = threadIdx.x; idx < 64 * 64; idx += 256) {
                int row = idx >> 6, c8 = idx & 63;
                *(uint4*)&Atile[row * 512 + c8 * 8] =
                    *(const uint4*)&Abf[(size_t)row * KTOT + kb * 512 + c8 * 8];
            }
#endif
        }
        __syncthreads();

        const unsigned short* wnt = Wp + (size_t)(n0 >> 4) * (KTOT / 32) * 512;
#pragma unroll 4
        for (int kc2 = 0; kc2 < 16; ++kc2) {
            int kc = kb * 16 + kc2;
            const unsigned short* bp = wnt + (size_t)kc * 512 + lane * 16;
            __builtin_prefetch(bp + 512, 0, 0);       // next chunk (L2-hot)
            v16bf bfrag = load_frag16(bp);
#pragma unroll
            for (int mt = 0; mt < 4; ++mt) {
                const unsigned short* ap =
                    &Atile[(mt * 16 + nloc) * 512 + kc2 * 32 + hiHalf * 16];
                v16bf afrag = load_frag16(ap);
#ifdef __HIP_DEVICE_COMPILE__
                acc[mt] = __builtin_amdgcn_wmma_f32_16x16x32_bf16(
                    false, afrag, false, bfrag, (short)0, acc[mt], false, false);
#endif
            }
        }
        __syncthreads();
    }

    // Epilogue: lane L, reg r -> row b = mt*16 + r + hiHalf*8, col n = n0+nloc
#pragma unroll
    for (int mt = 0; mt < 4; ++mt) {
#pragma unroll
        for (int rr = 0; rr < 8; ++rr) {
            int b = mt * 16 + rr + hiHalf * 8;
            int n = n0 + nloc;
            float val = acc[mt][rr] + bias[n];
            if (MODE == 0) {
                if (n < 512) {
                    qbuf[b * 512 + n] = val;
                } else if (n < 1024) {
                    int i = n - 512;
                    Kc[((((size_t)layer * BATCH + b) * HEADS + (i >> 6)) * TMAXC + pos) * HDIM
                       + (i & 63)] = f2bf(val);
                } else {
                    int i = n - 1024;
                    Vc[((((size_t)layer * BATCH + b) * HEADS + (i >> 6)) * TMAXC + pos) * HDIM
                       + (i & 63)] = f2bf(val);
                }
            } else if (MODE == 1) {
                hio[b * 512 + n] += val;
            } else {
                float g = 0.5f * val * (1.0f + erff(val * 0.70710678f));
                mout[(size_t)b * Ntot + n] = f2bf(g);
            }
        }
    }
}

// ---------------------------------------------------------------------------
// Attention: one wave per (b, head). fp32 softmax over bf16 KV cache.
// ---------------------------------------------------------------------------
__global__ __launch_bounds__(256) void attn_kernel(
    const float* __restrict__ qbuf,
    const unsigned short* __restrict__ Kc,
    const unsigned short* __restrict__ Vc,
    unsigned short* __restrict__ obuf, int layer, int pos) {
    __shared__ float qs[HEADS][HDIM];
    __shared__ float att[HEADS][TMAXC + 3];
    int b = blockIdx.x;
    int w = threadIdx.x >> 5;   // head
    int lane = threadIdx.x & 31;

    for (int d = lane; d < HDIM; d += 32)
        qs[w][d] = qbuf[b * DMODEL + w * HDIM + d];
    __syncthreads();

    const unsigned short* Kp = &Kc[(((size_t)layer * BATCH + b) * HEADS + w) * TMAXC * HDIM];
    const unsigned short* Vp = &Vc[(((size_t)layer * BATCH + b) * HEADS + w) * TMAXC * HDIM];

    float lmax = -3.4e38f;
    for (int base = 0; base <= pos; base += 32) {
        int tt = base + lane;
        float s = -3.4e38f;
        if (tt <= pos) {
            float a = 0.0f;
            const unsigned short* kr = Kp + (size_t)tt * HDIM;
#pragma unroll 8
            for (int d = 0; d < HDIM; ++d) a += qs[w][d] * bf2f(kr[d]);
            s = a * 0.125f;                  // 1/sqrt(64)
            att[w][tt] = s;
        }
        lmax = fmaxf(lmax, s);
    }
#pragma unroll
    for (int off = 16; off; off >>= 1) lmax = fmaxf(lmax, __shfl_xor(lmax, off, 32));
    __syncthreads();

    float lsum = 0.0f;
    for (int base = 0; base <= pos; base += 32) {
        int tt = base + lane;
        if (tt <= pos) {
            float e = __expf(att[w][tt] - lmax);
            att[w][tt] = e;
            lsum += e;
        }
    }
#pragma unroll
    for (int off = 16; off; off >>= 1) lsum += __shfl_xor(lsum, off, 32);
    __syncthreads();

    float inv = 1.0f / lsum;
    for (int d = lane; d < HDIM; d += 32) {
        float a = 0.0f;
        for (int tt = 0; tt <= pos; ++tt)
            a += att[w][tt] * bf2f(Vp[(size_t)tt * HDIM + d]);
        obuf[b * DMODEL + w * HDIM + d] = f2bf(a * inv);
    }
}

// ---------------------------------------------------------------------------
// Final LN + head GEMV + ODE state update (y, u)
// ---------------------------------------------------------------------------
__global__ __launch_bounds__(256) void head_kernel(
    const float* __restrict__ hbuf,
    const float* __restrict__ lnf_w, const float* __restrict__ lnf_b,
    const float* __restrict__ head_w, const float* __restrict__ head_b,
    const float* __restrict__ data,
    float* __restrict__ y, float* __restrict__ u) {
    __shared__ float red[8][4];
    int b = blockIdx.x;
    int wave = threadIdx.x >> 5, lane = threadIdx.x & 31;

    float s = 0.0f, q = 0.0f;
    for (int d = threadIdx.x; d < DMODEL; d += 256) {
        float v = hbuf[b * DMODEL + d];
        s += v; q += v * v;
    }
#pragma unroll
    for (int off = 16; off; off >>= 1) {
        s += __shfl_xor(s, off, 32);
        q += __shfl_xor(q, off, 32);
    }
    if (lane == 0) { red[wave][0] = s; red[wave][1] = q; }
    __syncthreads();
    if (threadIdx.x == 0) {
        float ss = 0.0f, qq = 0.0f;
        for (int i = 0; i < 8; ++i) { ss += red[i][0]; qq += red[i][1]; }
        float mu = ss * (1.0f / 512.0f);
        float var = qq * (1.0f / 512.0f) - mu * mu;
        red[0][2] = mu;
        red[0][3] = rsqrtf(var + 1e-5f);
    }
    __syncthreads();
    float mu = red[0][2], rs = red[0][3];

    float p0 = 0.0f, p1 = 0.0f;
    for (int d = threadIdx.x; d < DMODEL; d += 256) {
        float hf = (hbuf[b * DMODEL + d] - mu) * rs * lnf_w[d] + lnf_b[d];
        p0 += hf * head_w[d * NREF + 0];
        p1 += hf * head_w[d * NREF + 1];
    }
#pragma unroll
    for (int off = 16; off; off >>= 1) {
        p0 += __shfl_xor(p0, off, 32);
        p1 += __shfl_xor(p1, off, 32);
    }
    if (lane == 0) { red[wave][0] = p0; red[wave][1] = p1; }
    __syncthreads();
    if (threadIdx.x == 0) {
        float u0 = head_b[0], u1 = head_b[1];
        for (int i = 0; i < 8; ++i) { u0 += red[i][0]; u1 += red[i][1]; }
        float un[2] = { u0, u1 };
        for (int nr = 0; nr < 2; ++nr) {
            float ap = data[b * 4 + nr];
            float bp = data[b * 4 + 2 + nr];
            float yo = y[b * 2 + nr];
            y[b * 2 + nr] = yo + (-ap * yo + bp * un[nr]);
            u[b * 2 + nr] = un[nr];
        }
    }
}

// ---------------------------------------------------------------------------
extern "C" void kernel_launch(void* const* d_in, const int* in_sizes, int n_in,
                              void* d_out, int out_size, void* d_ws, size_t ws_size,
                              hipStream_t stream) {
    (void)in_sizes; (void)n_in; (void)out_size; (void)ws_size;
    const float* data    = (const float*)d_in[0];
    const float* r       = (const float*)d_in[1];
    const float* y_d     = (const float*)d_in[2];
    const float* wte_w   = (const float*)d_in[3];
    const float* wte_b   = (const float*)d_in[4];
    const float* wpe     = (const float*)d_in[5];
    const float* ln1_w   = (const float*)d_in[6];
    const float* ln1_b   = (const float*)d_in[7];
    const float* qkv_w   = (const float*)d_in[8];
    const float* qkv_b   = (const float*)d_in[9];
    const float* proj_w  = (const float*)d_in[10];
    const float* proj_b  = (const float*)d_in[11];
    const float* ln2_w   = (const float*)d_in[12];
    const float* ln2_b   = (const float*)d_in[13];
    const float* fc_w    = (const float*)d_in[14];
    const float* fc_b    = (const float*)d_in[15];
    const float* mproj_w = (const float*)d_in[16];
    const float* mproj_b = (const float*)d_in[17];
    const float* lnf_w   = (const float*)d_in[18];
    const float* lnf_b   = (const float*)d_in[19];
    const float* head_w  = (const float*)d_in[20];
    const float* head_b  = (const float*)d_in[21];
    float* Y = (float*)d_out;

    // Workspace carving (256B aligned)
    char* ws = (char*)d_ws;
    size_t off = 0;
    auto carve = [&](size_t bytes) -> char* {
        char* p = ws + off;
        off = (off + bytes + 255) & ~(size_t)255;
        return p;
    };
    unsigned short* WqkvP   = (unsigned short*)carve((size_t)LAYERS * 512 * 1536 * 2);
    unsigned short* WprojP  = (unsigned short*)carve((size_t)LAYERS * 512 * 512 * 2);
    unsigned short* WfcP    = (unsigned short*)carve((size_t)LAYERS * 512 * 2048 * 2);
    unsigned short* WmprojP = (unsigned short*)carve((size_t)LAYERS * 2048 * 512 * 2);
    unsigned short* Kc      = (unsigned short*)carve((size_t)LAYERS * BATCH * HEADS * TMAXC * HDIM * 2);
    unsigned short* Vc      = (unsigned short*)carve((size_t)LAYERS * BATCH * HEADS * TMAXC * HDIM * 2);
    float*          hbuf    = (float*)carve((size_t)BATCH * DMODEL * 4);
    float*          qbuf    = (float*)carve((size_t)BATCH * DMODEL * 4);
    unsigned short* obuf    = (unsigned short*)carve((size_t)BATCH * DMODEL * 2);
    unsigned short* mbuf    = (unsigned short*)carve((size_t)BATCH * 4 * DMODEL * 2);
    float*          yst     = (float*)carve((size_t)BATCH * NREF * 4);
    float*          ust     = (float*)carve((size_t)BATCH * NREF * 4);

    // One-time-per-call weight pack (bf16, WMMA-B layout; L2-resident after first use)
    pack_weights<<<2048, 256, 0, stream>>>(qkv_w,   WqkvP,   512, 1536);
    pack_weights<<<2048, 256, 0, stream>>>(proj_w,  WprojP,  512, 512);
    pack_weights<<<2048, 256, 0, stream>>>(fc_w,    WfcP,    512, 2048);
    pack_weights<<<2048, 256, 0, stream>>>(mproj_w, WmprojP, 2048, 512);
    init_state<<<1, 128, 0, stream>>>(y_d, yst, ust);

    for (int t = 0; t < TSTEPS; ++t) {
        embed_kernel<<<BATCH, 256, 0, stream>>>(r, wte_w, wte_b, wpe, yst, ust, hbuf, Y, t);
        for (int l = 0; l < LAYERS; ++l) {
            // LN1 + QKV GEMM (N=1536) with q/k/v scatter
            gemm64_wmma<512, 0, true><<<12, 256, 0, stream>>>(
                hbuf, nullptr, WqkvP + (size_t)l * 512 * 1536,
                qkv_b + (size_t)l * 1536, ln1_w + (size_t)l * DMODEL, ln1_b + (size_t)l * DMODEL,
                nullptr, qbuf, Kc, Vc, nullptr, l, t, 1536);
            attn_kernel<<<BATCH, 256, 0, stream>>>(qbuf, Kc, Vc, obuf, l, t);
            // proj GEMM (N=512), residual add (TDM-staged A)
            gemm64_wmma<512, 1, false><<<4, 256, 0, stream>>>(
                nullptr, obuf, WprojP + (size_t)l * 512 * 512,
                proj_b + (size_t)l * DMODEL, nullptr, nullptr,
                hbuf, nullptr, nullptr, nullptr, nullptr, l, t, 512);
            // LN2 + FC GEMM (N=2048), GELU
            gemm64_wmma<512, 2, true><<<16, 256, 0, stream>>>(
                hbuf, nullptr, WfcP + (size_t)l * 512 * 2048,
                fc_b + (size_t)l * 4 * DMODEL, ln2_w + (size_t)l * DMODEL, ln2_b + (size_t)l * DMODEL,
                nullptr, nullptr, nullptr, nullptr, mbuf, l, t, 2048);
            // mproj GEMM (K=2048, N=512), residual add (TDM-staged A)
            gemm64_wmma<2048, 1, false><<<4, 256, 0, stream>>>(
                nullptr, mbuf, WmprojP + (size_t)l * 2048 * 512,
                mproj_b + (size_t)l * DMODEL, nullptr, nullptr,
                hbuf, nullptr, nullptr, nullptr, nullptr, l, t, 512);
        }
        head_kernel<<<BATCH, 256, 0, stream>>>(hbuf, lnf_w, lnf_b, head_w, head_b, data, yst, ust);
    }
}
